// MambaBlock_44693429682705
// MI455X (gfx1250) — matrix operand order repaired and checked by reading, not compiled, not run on hardware
//
#include <hip/hip_runtime.h>
#include <hip/hip_bf16.h>

// ---------------------------------------------------------------------------
// MambaBlock for MI455X (gfx1250): bf16 WMMA GEMMs + fp32 scan + fused LN.
// B=4 T=2048 D=512 E=1024 S=16 ; M = B*T = 8192
// Wave tile 32x64 (2 M-frags x 4 N-frags), fragment loads batched per K-chunk
// so WMMAs overlap L2 latency instead of s_wait_loadcnt 0x0 per MMA.
// ---------------------------------------------------------------------------

typedef __attribute__((ext_vector_type(16))) __bf16 v16bf;
typedef __attribute__((ext_vector_type(8)))  __bf16 v8bf;
typedef __attribute__((ext_vector_type(8)))  float  v8f;

#define BB 4
#define TT 2048
#define DD 512
#define EE 1024
#define SS 16
#define MM (BB * TT)        // 8192 rows

__device__ __forceinline__ __bf16 f2bf(float f) {
  union { float f; unsigned int u; } a; a.f = f;
  unsigned int r = a.u + 0x7FFFu + ((a.u >> 16) & 1u);   // round-to-nearest-even
  unsigned short h = (unsigned short)(r >> 16);
  __bf16 b;
  __builtin_memcpy(&b, &h, sizeof(h));
  return b;
}

union ufrag { v16bf v; v8bf h[2]; };

// -------------------- pack / transpose kernels -----------------------------

__global__ __launch_bounds__(256) void cvt_bf16_kernel(
    const float* __restrict__ in, __bf16* __restrict__ out, int n) {
  int i = blockIdx.x * 256 + threadIdx.x;
  if (i < n) out[i] = f2bf(in[i]);
}

// in: [R][C] f32 row-major  ->  out: [C][R] bf16 row-major
__global__ __launch_bounds__(256) void transpose_bf16_kernel(
    const float* __restrict__ in, __bf16* __restrict__ out, int R, int C) {
  int i = blockIdx.x * 256 + threadIdx.x;
  if (i < R * C) {
    int r = i / C, c = i % C;
    out[(size_t)c * R + r] = f2bf(in[i]);
  }
}

// -------------------- GEMM1: u / gate  (M=8192,K=512,N=1024) ---------------
// A  : xb   [M][512]  bf16 row-major
// BT : W^T  [1024][512] bf16 (so B[k][n] == BT[n][k])
// out: f32 [M][1024], epilogue bias (+ optional sigmoid)
// WG: 256 threads = 8 waves (2 M x 4 N); WG tile 64(M) x 256(N);
// wave tile 32 x 64: acc[2][4], A frags reused 4x, B frags reused 2x.

__global__ __launch_bounds__(256) void gemm_in_kernel(
    const __bf16* __restrict__ A, const __bf16* __restrict__ BT,
    const float* __restrict__ bias, float* __restrict__ out, int applySigmoid) {
  constexpr int K = DD;    // 512
  constexpr int N = EE;    // 1024
  __shared__ __align__(16) __bf16 As[64][64];

  const int tid  = threadIdx.x;
  const int lane = tid & 31;
  const int w    = tid >> 5;     // 0..7
  const int wm   = w >> 2;       // 0..1 -> 32-row half
  const int wn   = w & 3;        // 0..3 -> 64-col slice
  const int Mblock = blockIdx.x * 64;
  const int Nbase  = blockIdx.y * 256 + wn * 64;

  const int col = lane & 15;
  const int kbA = (lane < 16) ? 0 : 16;   // A frag contiguous k..k+15
  const int kbB = (lane < 16) ? 0 : 16;   // B frag contiguous k..k+15

  v8f acc[2][4] = {};

  for (int kc = 0; kc < K; kc += 64) {
    // cooperative A stage: 64 rows x 64 k, two b128 per thread
    {
      int r   = tid >> 2;          // 0..63
      int c16 = (tid & 3) * 16;    // 0,16,32,48
      const __bf16* srcrow = A + (size_t)(Mblock + r) * K + kc + c16;
      *(v8bf*)(&As[r][c16])     = *(const v8bf*)(srcrow);
      *(v8bf*)(&As[r][c16 + 8]) = *(const v8bf*)(srcrow + 8);
    }
    __syncthreads();

    // batch ALL fragment loads for this K-chunk before the MMA block
    ufrag af[2][2];                 // [ks][mt]
#pragma unroll
    for (int ks2 = 0; ks2 < 2; ks2++)
#pragma unroll
      for (int mt = 0; mt < 2; mt++) {
        const __bf16* ap = &As[wm * 32 + mt * 16 + col][ks2 * 32 + kbA];
        af[ks2][mt].h[0] = *(const v8bf*)(ap);
        af[ks2][mt].h[1] = *(const v8bf*)(ap + 8);
      }
    v16bf bv[2][4];                 // [ks][nt]
#pragma unroll
    for (int ks2 = 0; ks2 < 2; ks2++)
#pragma unroll
      for (int nt = 0; nt < 4; nt++) {
        int n = Nbase + nt * 16 + col;
        bv[ks2][nt] =
            *(const v16bf*)(BT + (size_t)n * K + kc + ks2 * 32 + kbB);
      }
    // 16 WMMAs consuming the batched fragments
#pragma unroll
    for (int ks2 = 0; ks2 < 2; ks2++)
#pragma unroll
      for (int mt = 0; mt < 2; mt++)
#pragma unroll
        for (int nt = 0; nt < 4; nt++)
          acc[mt][nt] = __builtin_amdgcn_wmma_f32_16x16x32_bf16(
              false, af[ks2][mt].v, false, bv[ks2][nt], (short)0,
              acc[mt][nt], false, false);
    __syncthreads();
  }

  const int rhalf = (lane < 16) ? 0 : 8;
#pragma unroll
  for (int mt = 0; mt < 2; mt++) {
    int rbase = Mblock + wm * 32 + mt * 16 + rhalf;
#pragma unroll
    for (int nt = 0; nt < 4; nt++) {
      int n = Nbase + nt * 16 + col;
      float bs = bias[n];
#pragma unroll
      for (int r = 0; r < 8; r++) {
        float v = acc[mt][nt][r] + bs;
        if (applySigmoid) v = 1.0f / (1.0f + __expf(-v));
        out[(size_t)(rbase + r) * N + n] = v;
      }
    }
  }
}

// -------------------- sequential SSM scan ----------------------------------
// One lane per (b,e); 16 fp32 states in registers; coalesced over e.

__global__ __launch_bounds__(256) void scan_kernel(
    const float* __restrict__ u, const float* __restrict__ gate,
    const float* __restrict__ A_log, const float* __restrict__ Bm,
    const float* __restrict__ Cm, const float* __restrict__ dt,
    __bf16* __restrict__ yg) {
  int tid = blockIdx.x * 256 + threadIdx.x;   // 0..4095
  int b = tid >> 10;
  int e = tid & (EE - 1);
  float d = dt[e];

  float ad[SS], bm[SS], cm[SS], h[SS];
#pragma unroll
  for (int s = 0; s < SS; s++) {
    ad[s] = __expf(A_log[e * SS + s] * d);
    bm[s] = Bm[e * SS + s];
    cm[s] = Cm[e * SS + s];
    h[s]  = 0.0f;
  }

  const float* up = u    + (size_t)b * TT * EE + e;
  const float* gp = gate + (size_t)b * TT * EE + e;
  __bf16*      yp = yg   + (size_t)b * TT * EE + e;

  for (int t = 0; t < TT; t++) {
    if (t + 32 < TT) {
      __builtin_prefetch(up + (size_t)(t + 32) * EE, 0, 0);
      __builtin_prefetch(gp + (size_t)(t + 32) * EE, 0, 0);
    }
    float ut = up[(size_t)t * EE];
    float g  = gp[(size_t)t * EE];
    float y = 0.0f;
#pragma unroll
    for (int s = 0; s < SS; s++) {
      h[s] = ad[s] * h[s] + bm[s] * ut;
      y += h[s] * cm[s];
    }
    yp[(size_t)t * EE] = f2bf(y * g);
  }
}

// -------------------- GEMM2 + residual + LayerNorm -------------------------
// A  : ygb [M][1024] bf16 ; BT : W_out^T [512][1024] bf16
// WG: 256 threads = 8 waves; WG tile 32(M) x 512(all N) so LN fuses.
// Each wave: all 32 rows x 64 cols -> acc[2][4].

__global__ __launch_bounds__(256) void gemm_out_ln_kernel(
    const __bf16* __restrict__ A, const __bf16* __restrict__ BT,
    const float* __restrict__ bias, const float* __restrict__ xres,
    const float* __restrict__ gamma, const float* __restrict__ beta,
    float* __restrict__ out) {
  constexpr int K = EE;   // 1024
  constexpr int N = DD;   // 512
  __shared__ __align__(16) __bf16 As[32][64];
  __shared__ float Os[32][DD];

  const int tid  = threadIdx.x;
  const int lane = tid & 31;
  const int w    = tid >> 5;        // 0..7, owns cols [w*64, w*64+64)
  const int Mblock = blockIdx.x * 32;

  const int col = lane & 15;
  const int kbA = (lane < 16) ? 0 : 16;
  const int kbB = (lane < 16) ? 0 : 16;

  v8f acc[2][4] = {};

  for (int kc = 0; kc < K; kc += 64) {
    {
      int r  = tid >> 3;            // 0..31
      int c8 = (tid & 7) * 8;       // 0..56
      *(v8bf*)(&As[r][c8]) =
          *(const v8bf*)(A + (size_t)(Mblock + r) * K + kc + c8);
    }
    __syncthreads();

    ufrag af[2][2];                 // [ks][mt] : rows mt*16..
#pragma unroll
    for (int ks2 = 0; ks2 < 2; ks2++)
#pragma unroll
      for (int mt = 0; mt < 2; mt++) {
        const __bf16* ap = &As[mt * 16 + col][ks2 * 32 + kbA];
        af[ks2][mt].h[0] = *(const v8bf*)(ap);
        af[ks2][mt].h[1] = *(const v8bf*)(ap + 8);
      }
    v16bf bv[2][4];
#pragma unroll
    for (int ks2 = 0; ks2 < 2; ks2++)
#pragma unroll
      for (int nt = 0; nt < 4; nt++) {
        int n = w * 64 + nt * 16 + col;
        bv[ks2][nt] =
            *(const v16bf*)(BT + (size_t)n * K + kc + ks2 * 32 + kbB);
      }
#pragma unroll
    for (int ks2 = 0; ks2 < 2; ks2++)
#pragma unroll
      for (int mt = 0; mt < 2; mt++)
#pragma unroll
        for (int nt = 0; nt < 4; nt++)
          acc[mt][nt] = __builtin_amdgcn_wmma_f32_16x16x32_bf16(
              false, af[ks2][mt].v, false, bv[ks2][nt], (short)0,
              acc[mt][nt], false, false);
    __syncthreads();
  }

  // spill accumulators (+bias) to LDS for the row-wise LayerNorm
  const int rhalf = (lane < 16) ? 0 : 8;
#pragma unroll
  for (int mt = 0; mt < 2; mt++)
#pragma unroll
    for (int nt = 0; nt < 4; nt++) {
      int n = w * 64 + nt * 16 + col;
      float bs = bias[n];
#pragma unroll
      for (int r = 0; r < 8; r++)
        Os[mt * 16 + rhalf + r][n] = acc[mt][nt][r] + bs;
    }
  __syncthreads();

  // LayerNorm: 8 waves x 4 rows; residual add from x; wave32 reductions
#pragma unroll
  for (int rr = 0; rr < 4; rr++) {
    int row = w * 4 + rr;
    const float* xp = xres + (size_t)(Mblock + row) * N;
    float vals[16];
    float s = 0.0f, s2 = 0.0f;
#pragma unroll
    for (int i = 0; i < 16; i++) {
      int c = lane + i * 32;
      float v = Os[row][c] + xp[c];
      vals[i] = v;
      s += v; s2 += v * v;
    }
#pragma unroll
    for (int off = 16; off > 0; off >>= 1) {
      s  += __shfl_xor(s,  off, 32);
      s2 += __shfl_xor(s2, off, 32);
    }
    float mu  = s * (1.0f / (float)N);
    float var = s2 * (1.0f / (float)N) - mu * mu;
    float inv = rsqrtf(var + 1e-5f);
    float* op = out + (size_t)(Mblock + row) * N;
#pragma unroll
    for (int i = 0; i < 16; i++) {
      int c = lane + i * 32;
      op[c] = (vals[i] - mu) * inv * gamma[c] + beta[c];
    }
  }
}

// -------------------- launch -----------------------------------------------

extern "C" void kernel_launch(void* const* d_in, const int* in_sizes, int n_in,
                              void* d_out, int out_size, void* d_ws, size_t ws_size,
                              hipStream_t stream) {
  (void)in_sizes; (void)n_in; (void)out_size; (void)ws_size;

  const float* x      = (const float*)d_in[0];
  const float* W_in   = (const float*)d_in[1];
  const float* b_in   = (const float*)d_in[2];
  const float* W_gate = (const float*)d_in[3];
  const float* b_gate = (const float*)d_in[4];
  const float* A_log  = (const float*)d_in[5];
  const float* Bm     = (const float*)d_in[6];
  const float* Cm     = (const float*)d_in[7];
  const float* dt     = (const float*)d_in[8];
  const float* W_out  = (const float*)d_in[9];
  const float* b_out  = (const float*)d_in[10];
  const float* gamma  = (const float*)d_in[11];
  const float* beta   = (const float*)d_in[12];
  float* out = (float*)d_out;

  // workspace layout (bytes)
  constexpr size_t XB_OFF   = 0;                              // 8192*512  bf16
  constexpr size_t WTI_OFF  = XB_OFF  + (size_t)MM * DD * 2;  // [1024][512] bf16
  constexpr size_t WTG_OFF  = WTI_OFF + (size_t)DD * EE * 2;
  constexpr size_t WTO_OFF  = WTG_OFF + (size_t)DD * EE * 2;  // [512][1024] bf16
  constexpr size_t U_OFF    = WTO_OFF + (size_t)EE * DD * 2;  // [M][E] f32
  constexpr size_t G_OFF    = U_OFF   + (size_t)MM * EE * 4;  // [M][E] f32
  constexpr size_t YG_OFF   = G_OFF   + (size_t)MM * EE * 4;  // [M][E] bf16

  char* ws = (char*)d_ws;
  __bf16* xb  = (__bf16*)(ws + XB_OFF);
  __bf16* WiT = (__bf16*)(ws + WTI_OFF);
  __bf16* WgT = (__bf16*)(ws + WTG_OFF);
  __bf16* WoT = (__bf16*)(ws + WTO_OFF);
  float*  u   = (float*)(ws + U_OFF);
  float*  g   = (float*)(ws + G_OFF);
  __bf16* ygb = (__bf16*)(ws + YG_OFF);

  // 1. pack x -> bf16 ; transpose weights -> bf16 [N][K]
  {
    int n = MM * DD;
    cvt_bf16_kernel<<<(n + 255) / 256, 256, 0, stream>>>(x, xb, n);
  }
  transpose_bf16_kernel<<<(DD * EE + 255) / 256, 256, 0, stream>>>(W_in,   WiT, DD, EE);
  transpose_bf16_kernel<<<(DD * EE + 255) / 256, 256, 0, stream>>>(W_gate, WgT, DD, EE);
  transpose_bf16_kernel<<<(EE * DD + 255) / 256, 256, 0, stream>>>(W_out,  WoT, EE, DD);

  // 2. input GEMMs: u = x@W_in+b_in ; gate = sigmoid(x@W_gate+b_gate)
  {
    dim3 grid(MM / 64, EE / 256);
    gemm_in_kernel<<<grid, 256, 0, stream>>>(xb, WiT, b_in,   u, 0);
    gemm_in_kernel<<<grid, 256, 0, stream>>>(xb, WgT, b_gate, g, 1);
  }

  // 3. sequential scan, gated output -> bf16
  scan_kernel<<<(BB * EE) / 256, 256, 0, stream>>>(u, g, A_log, Bm, Cm, dt, ygb);

  // 4. output GEMM + residual + LayerNorm (writes d_out)
  gemm_out_ln_kernel<<<MM / 32, 256, 0, stream>>>(ygb, WoT, b_out, x, gamma, beta, out);
}
